// NetworkRetrain_8340826489130
// MI455X (gfx1250) — matrix operand order repaired and checked by reading, not compiled, not run on hardware
//
#include <hip/hip_runtime.h>

// ---------------------------------------------------------------------------
// MI455X (gfx1250) implementation.
// Roofline: 243 MB of f32 weights dominate traffic (~10.4 us @ 23.3 TB/s);
// 15.6 GFLOP total. BF16 WMMA (f32 accum) keeps compute under the HBM roof.
// M=128 fits one 8-wave block -> every weight byte is streamed from HBM
// exactly once, converted f32->bf16 in flight into a transposed LDS tile.
// K-loop is unrolled by 2 with two A-fragment register sets (no rotate copy,
// no WMMA->VALU hazard nops) and compile-time LDS buffer parity; B fragments
// use 4 independent register sets so the 4 WMMAs pipeline behind partial
// s_wait_dscnt waits.
// ---------------------------------------------------------------------------

typedef __attribute__((ext_vector_type(16))) __bf16 v16bf;
typedef __attribute__((ext_vector_type(8)))  float  v8f;

union Frag32B { uint4 u[2]; v16bf v; };

struct EdgeParams {
  const __bf16* X[6];   // per-edge input activations [128 x K] bf16 row-major
  const float*  W[6];   // per-edge weights [K x N] f32 row-major
  const float*  B[6];   // per-edge bias [N] f32
  int K[6];
  int op[6];            // 0 = linear, 1 = relu, 2 = lrelu(0.2)
  int N;
};

__device__ __forceinline__ unsigned int f2bf_bits(float f) {
  __bf16 h = (__bf16)f;
  return (unsigned int)__builtin_bit_cast(unsigned short, h);
}

__device__ __forceinline__ v8f wmma_bf16(const Frag32B& a, const Frag32B& b,
                                         v8f c) {
  return __builtin_amdgcn_wmma_f32_16x16x32_bf16(
      false, a.v, false, b.v, (short)0, c, false, false);
}

// ---------------------------------------------------------------------------
// Batched per-edge GEMM: Y[edge] = act(X_edge @ W_edge + b_edge), bf16 out.
// Block: 256 threads = 8 wave32. Tile: M=128 (full batch), N=64, K-step 32.
// ---------------------------------------------------------------------------
__global__ __launch_bounds__(256)
void gemm_edges_kernel(EdgeParams p, __bf16* __restrict__ Y) {
  const int edge = blockIdx.y;
  const int N    = p.N;
  const int K    = p.K[edge];
  const int nb   = blockIdx.x * 64;
  const __bf16* __restrict__ X = p.X[edge];
  const float*  __restrict__ W = p.W[edge];

  const int tid  = threadIdx.x;
  const int lane = tid & 31;
  const int wave = tid >> 5;
  const int half = lane >> 4;   // which 16-lane half of the wave
  const int l16  = lane & 15;

  // Transposed bf16 W tile: 64 cols x 32 K-halves, padded to 40 halves/col
  // (stride 80 B -> 16B-aligned b128 reads, conflict-free bank pattern).
  __shared__ __align__(16) unsigned int Wt[2][64 * 20];

  v8f zero = {};
  v8f acc0 = zero, acc1 = zero, acc2 = zero, acc3 = zero;

  // Cooperative W-tile loader: thread -> 2 K-rows x 4 N-cols (float4 pairs),
  // converted to bf16 and written transposed (packed pairs along K).
  const int rp   = tid >> 4;        // 0..15 -> K rows 2rp, 2rp+1
  const int cg   = tid & 15;        // 0..15 -> cols 4cg..4cg+3
  const int krow = rp << 1;
  const int col0 = cg << 2;

  auto load_tile = [&](int kc, int buf) {
    const float* w0 = W + (size_t)(kc + krow) * N + (nb + col0);
    const float* w1 = w0 + N;
    float4 ra = *(const float4*)w0;
    float4 rb = *(const float4*)w1;
    const float* pa = &ra.x;
    const float* pb = &rb.x;
    unsigned int* dst = &Wt[buf][0];
    #pragma unroll
    for (int j = 0; j < 4; ++j) {
      unsigned int lo = f2bf_bits(pa[j]);
      unsigned int hi = f2bf_bits(pb[j]);
      dst[(col0 + j) * 20 + rp] = lo | (hi << 16);
    }
  };

  const int m = (wave << 4) + l16;           // batch row 0..127
  const __bf16* __restrict__ Xrow = X + (size_t)m * K;

  // A layout (16x32 bf16): elems 0..7 = K kc+half*16+(0..7),
  //                        elems 8..15 = K kc+half*16+8+(0..7).
  auto load_a = [&](Frag32B& a, int kc) {
    a.u[0] = *(const uint4*)(Xrow + kc + half * 16);
    a.u[1] = *(const uint4*)(Xrow + kc + half * 16 + 8);
  };

  // 4 B fragments (32x16 bf16: col = g*16+l16, K = half*16+(0..15),
  // contiguous in the transposed tile) -> 8 ds_load_b128, then 4 WMMAs.
  auto compute = [&](int buf, const Frag32B& a) {
    const unsigned int* base = &Wt[buf][half * 8];
    Frag32B b0, b1, b2, b3;
    b0.u[0] = *(const uint4*)&base[(l16 +  0) * 20];
    b0.u[1] = *(const uint4*)&base[(l16 +  0) * 20 + 4];
    b1.u[0] = *(const uint4*)&base[(l16 + 16) * 20];
    b1.u[1] = *(const uint4*)&base[(l16 + 16) * 20 + 4];
    b2.u[0] = *(const uint4*)&base[(l16 + 32) * 20];
    b2.u[1] = *(const uint4*)&base[(l16 + 32) * 20 + 4];
    b3.u[0] = *(const uint4*)&base[(l16 + 48) * 20];
    b3.u[1] = *(const uint4*)&base[(l16 + 48) * 20 + 4];
    acc0 = wmma_bf16(a, b0, acc0);
    acc1 = wmma_bf16(a, b1, acc1);
    acc2 = wmma_bf16(a, b2, acc2);
    acc3 = wmma_bf16(a, b3, acc3);
  };

  const int nchunks = K >> 5;   // K in {512..4096} -> nchunks even (16..128)

  // Prologue: tile 0 into LDS buf 0, A fragment for chunk 0.
  load_tile(0, 0);
  Frag32B a0, a1;
  load_a(a0, 0);
  __syncthreads();

  for (int c = 0; c < nchunks; c += 2) {
    const int kc = c << 5;

    // Even chunk: stage odd tile (buf 1) + its A fragment, compute buf 0.
    load_tile(kc + 32, 1);
    load_a(a1, kc + 32);
    if (c + 2 < nchunks)
      __builtin_prefetch(W + (size_t)(kc + 64) * N + nb, 0, 0);
    compute(0, a0);
    __syncthreads();

    // Odd chunk: stage next even tile (buf 0) + its A, compute buf 1.
    if (c + 2 < nchunks) {
      load_tile(kc + 64, 0);
      load_a(a0, kc + 64);
      if (c + 3 < nchunks)
        __builtin_prefetch(W + (size_t)(kc + 96) * N + nb, 0, 0);
    }
    compute(1, a1);
    __syncthreads();
  }

  // Epilogue: bias + activation, bf16 store.
  // C/D layout: vgpr i, lanes<16 -> M=i, lanes>=16 -> M=8+i; N = lane&15.
  const float* __restrict__ bias = p.B[edge] + nb;
  const int op = p.op[edge];
  __bf16* __restrict__ Yout = Y + (size_t)edge * 128 * N + nb;
  v8f accs[4] = {acc0, acc1, acc2, acc3};
  #pragma unroll
  for (int g = 0; g < 4; ++g) {
    const int n  = (g << 4) + l16;
    const float bv = bias[n];
    #pragma unroll
    for (int i = 0; i < 8; ++i) {
      float y = accs[g][i] + bv;
      if (op == 1)      y = fmaxf(y, 0.f);
      else if (op == 2) y = (y > 0.f) ? y : 0.2f * y;
      const int row = (wave << 4) + i + half * 8;
      Yout[(size_t)row * N + n] = (__bf16)y;
    }
  }
}

// ---------------------------------------------------------------------------
// SE gate: avg over F per (b,c); h=relu(avg@W1+b1) (r=1);
// G[b][c] = sigmoid(h*W2[c]+b2[c]) * agg[c].  One block per batch row.
// ---------------------------------------------------------------------------
__global__ __launch_bounds__(256)
void gate_kernel(const __bf16* __restrict__ Y, int C, int N,
                 const float* __restrict__ W1, const float* __restrict__ b1,
                 const float* __restrict__ W2, const float* __restrict__ b2,
                 const float* __restrict__ agg, float* __restrict__ G) {
  const int b = blockIdx.x;
  __shared__ float red[256];
  __shared__ float avg[8];
  for (int c = 0; c < C; ++c) {
    const __bf16* row = Y + ((size_t)c * 128 + b) * N;
    float s = 0.f;
    for (int f = threadIdx.x; f < N; f += 256) s += (float)row[f];
    red[threadIdx.x] = s;
    __syncthreads();
    for (int off = 128; off > 0; off >>= 1) {
      if (threadIdx.x < off) red[threadIdx.x] += red[threadIdx.x + off];
      __syncthreads();
    }
    if (threadIdx.x == 0) avg[c] = red[0] / (float)N;
    __syncthreads();
  }
  if (threadIdx.x < C) {
    float h = b1[0];
    for (int c = 0; c < C; ++c) h += avg[c] * W1[c];
    h = fmaxf(h, 0.f);
    const float z = h * W2[threadIdx.x] + b2[threadIdx.x];
    const float w = 1.f / (1.f + __expf(-z));
    G[b * 8 + threadIdx.x] = w * agg[threadIdx.x];
  }
}

// ---------------------------------------------------------------------------
// Gated channel sum. i<3: bf16 state out. Final node: plain sum -> f32 d_out.
// Grid: (N/256, 128).
// ---------------------------------------------------------------------------
__global__ __launch_bounds__(256)
void combine_kernel(const __bf16* __restrict__ Y, const float* __restrict__ G,
                    int C, int N, __bf16* __restrict__ Obf,
                    float* __restrict__ Of) {
  const int b = blockIdx.y;
  const int f = blockIdx.x * 256 + threadIdx.x;
  float s = 0.f;
  for (int c = 0; c < C; ++c) {
    const float w = G ? G[b * 8 + c] : 1.f;
    s += (float)Y[((size_t)c * 128 + b) * N + f] * w;
  }
  const size_t idx = (size_t)b * N + f;
  if (Of) Of[idx] = s;
  else    Obf[idx] = (__bf16)s;
}

// f32 inputs -> bf16 activations; X2 = [s0 | s1].
__global__ __launch_bounds__(256)
void convert_kernel(const float* __restrict__ s0, const float* __restrict__ s1,
                    __bf16* __restrict__ X0, __bf16* __restrict__ X1,
                    __bf16* __restrict__ X2) {
  const int idx = blockIdx.x * 256 + threadIdx.x;   // 128*1024
  const int b = idx >> 10, f = idx & 1023;
  const float a0 = s0[idx], a1 = s1[idx];
  X0[idx] = (__bf16)a0;
  X1[idx] = (__bf16)a1;
  X2[(size_t)b * 2048 + f]        = (__bf16)a0;
  X2[(size_t)b * 2048 + 1024 + f] = (__bf16)a1;
}

extern "C" void kernel_launch(void* const* d_in, const int* in_sizes, int n_in,
                              void* d_out, int out_size, void* d_ws, size_t ws_size,
                              hipStream_t stream) {
  (void)in_sizes; (void)n_in; (void)out_size; (void)ws_size;
  // setup_inputs order: s_1, s_0, Ws[0..17], bs[0..17], ca_params[0..2]x{W1,b1,W2,b2,agg}
  const float* s1 = (const float*)d_in[0];
  const float* s0 = (const float*)d_in[1];

  char* ws = (char*)d_ws;
  __bf16* X0 = (__bf16*)(ws + 0);          // 128x1024 bf16
  __bf16* X1 = (__bf16*)(ws + 262144);     // 128x1024
  __bf16* X2 = (__bf16*)(ws + 524288);     // 128x2048
  __bf16* S0 = (__bf16*)(ws + 1048576);    // 128x512
  __bf16* S1 = (__bf16*)(ws + 1179648);    // 128x2048
  __bf16* S2 = (__bf16*)(ws + 1703936);    // 128x4096
  float*  G  = (float*) (ws + 2752512);    // 128x8
  __bf16* Yb = (__bf16*)(ws + 2756608);    // up to 5x128x4096 bf16

  convert_kernel<<<512, 256, 0, stream>>>(s0, s1, X0, X1, X2);

  const __bf16* states[6] = {X0, X1, X2, S0, S1, S2};
  __bf16*       nodeOut[3] = {S0, S1, S2};
  const int stateK[6] = {1024, 1024, 2048, 512, 2048, 4096};
  const int nodeN[4]  = {512, 2048, 4096, 2048};

  int e = 0;
  for (int i = 0; i < 4; ++i) {
    const int C = i + 3;
    EdgeParams p = {};
    p.N = nodeN[i];
    for (int j = 0; j < C; ++j) {
      p.X[j]  = states[j];
      p.K[j]  = stateK[j];
      p.W[j]  = (const float*)d_in[2 + e + j];
      p.B[j]  = (const float*)d_in[20 + e + j];
      p.op[j] = (e + j) % 3;
    }
    gemm_edges_kernel<<<dim3(p.N / 64, C), 256, 0, stream>>>(p, Yb);

    if (i < 3) {
      const int base = 38 + 5 * i;
      gate_kernel<<<128, 256, 0, stream>>>(
          Yb, C, p.N,
          (const float*)d_in[base + 0], (const float*)d_in[base + 1],
          (const float*)d_in[base + 2], (const float*)d_in[base + 3],
          (const float*)d_in[base + 4], G);
      combine_kernel<<<dim3(p.N / 256, 128), 256, 0, stream>>>(
          Yb, G, C, p.N, nodeOut[i], nullptr);
    } else {
      combine_kernel<<<dim3(p.N / 256, 128), 256, 0, stream>>>(
          Yb, nullptr, C, p.N, nullptr, (float*)d_out);
    }
    e += C;
  }
}
